// CausalSelfAttention_58652073394249
// MI455X (gfx1250) — matrix operand order repaired and checked by reading, not compiled, not run on hardware
//
#include <hip/hip_runtime.h>
#include <hip/hip_bf16.h>

// ---- problem constants (shared host/device) ----
constexpr int Bn = 4, Tn = 2048, Cn = 512, Hn = 8, Dn = 64;
constexpr int MW = Tn / 32;  // mask words per row = 64

typedef __bf16 bf16;
typedef __bf16 v16bf __attribute__((ext_vector_type(16)));
typedef float  v8f   __attribute__((ext_vector_type(8)));

union Frag { v16bf bf; uint4 u[2]; };

// A-operand 16x32 bf16 (row-major source, ld elements/row):
// lane: m=lane&15, half=lane>>4.  V0-3 <- K=half*8..+7, V4-7 <- K=16+half*8..+7
__device__ __forceinline__ void load_a_frag(Frag& f, const bf16* src, int row0, int ld,
                                            int k0, int lane) {
  int m = lane & 15, h = lane >> 4;
  const bf16* p = src + (size_t)(row0 + m) * ld + k0 + h * 8;
  f.u[0] = *(const uint4*)p;
  f.u[1] = *(const uint4*)(p + 16);
}

// B-operand 32x16 bf16 from "N-major" source: element B[k][n] = src[(n0+n)*ld + k0 + k]
// lane: n=lane&15, kh=lane>>4.  V0-3 <- K=kh*8..+7, V4-7 <- K=16+kh*8..+7
__device__ __forceinline__ void load_b_frag(Frag& f, const bf16* src, int n0, int ld,
                                            int k0, int lane) {
  int n = lane & 15, kh = lane >> 4;
  const bf16* p = src + (size_t)(n0 + n) * ld + k0 + kh * 8;
  f.u[0] = *(const uint4*)p;
  f.u[1] = *(const uint4*)(p + 16);
}

__device__ __forceinline__ v8f wmma_bf16(const Frag& a, const Frag& b, v8f c) {
  return __builtin_amdgcn_wmma_f32_16x16x32_bf16(false, a.bf, false, b.bf,
                                                 (short)0, c, false, false);
}

// ---------------- prep kernels ----------------
__global__ void cvt_f32_bf16_kernel(const float* __restrict__ in, bf16* __restrict__ out, int n) {
  int i = blockIdx.x * blockDim.x + threadIdx.x;
  if (i < n) out[i] = (bf16)in[i];
}

// w[K][N] (f32) -> wT[N][K] (bf16)
__global__ void transpose_w_kernel(const float* __restrict__ w, bf16* __restrict__ wT,
                                   int K, int N) {
  int i = blockIdx.x * blockDim.x + threadIdx.x;
  if (i >= K * N) return;
  int n = i / K, k = i % K;
  wT[i] = (bf16)w[(size_t)k * N + n];
}

// adj bytes (0/1) -> bit words, 32 keys per word
__global__ void pack_mask_kernel(const unsigned char* __restrict__ adj,
                                 unsigned int* __restrict__ mb, int nwords) {
  int i = blockIdx.x * blockDim.x + threadIdx.x;
  if (i >= nwords) return;
  const unsigned long long* p = (const unsigned long long*)(adj + (size_t)i * 32);
  unsigned int w = 0;
#pragma unroll
  for (int j = 0; j < 4; ++j) {
    unsigned long long u = p[j];
#pragma unroll
    for (int k = 0; k < 8; ++k)
      w |= ((unsigned int)(u >> (8 * k)) & 1u) << (j * 8 + k);
  }
  mb[i] = w;
}

// ---------------- QKV projection ----------------
// 16x64 tile per wave: A fragment reused across 4 B fragments (2.5 b128 loads / WMMA).
// xb [B*T][C] bf16 @ wqkvT [3C][C] bf16 -> Q[B,H,T,D]*1/8, K[B,H,T,D], Vt[B,H,D,T]
__global__ void __launch_bounds__(256) qkv_gemm_kernel(
    const bf16* __restrict__ xb, const bf16* __restrict__ wT,
    const float* __restrict__ bias, bf16* __restrict__ Q, bf16* __restrict__ Kd,
    bf16* __restrict__ Vt) {
  int wave = (blockIdx.x * blockDim.x + threadIdx.x) >> 5;
  int lane = threadIdx.x & 31;
  const int NT = (3 * Cn) / 64;  // 24 column tiles of 64
  int tm = wave / NT, tn = wave % NT;
  if (tm >= (Bn * Tn) / 16) return;

  v8f acc[4] = {{}, {}, {}, {}};
  for (int k0 = 0; k0 < Cn; k0 += 32) {
    Frag a;
    load_a_frag(a, xb, tm * 16, Cn, k0, lane);
#pragma unroll
    for (int dt = 0; dt < 4; ++dt) {
      Frag bf;
      load_b_frag(bf, wT, tn * 64 + dt * 16, Cn, k0, lane);
      acc[dt] = wmma_bf16(a, bf, acc[dt]);
    }
  }

  int n = lane & 15, hf = lane >> 4;
#pragma unroll
  for (int dt = 0; dt < 4; ++dt) {
    int col = tn * 64 + dt * 16 + n;
    float bv = bias[col];
    // whole 16-col subtile is in one of Q/K/V and one head (64-aligned blocks)
    int cls = col >> 9;          // 0=Q 1=K 2=V
    int c2 = col & (Cn - 1);
    int hh = c2 >> 6, d = c2 & 63;
#pragma unroll
    for (int r = 0; r < 8; ++r) {
      int rowg = tm * 16 + r + 8 * hf;
      int b = rowg >> 11, t = rowg & (Tn - 1);
      float v = acc[dt][r] + bv;
      if (cls == 0) {
        Q[(((size_t)(b * Hn + hh)) * Tn + t) * Dn + d] = (bf16)(v * 0.125f);
      } else if (cls == 1) {
        Kd[(((size_t)(b * Hn + hh)) * Tn + t) * Dn + d] = (bf16)v;
      } else {
        Vt[(((size_t)(b * Hn + hh)) * Dn + d) * Tn + t] = (bf16)v;
      }
    }
  }
}

// ---------------- flash attention ----------------
__global__ void __launch_bounds__(128) attn_kernel(
    const bf16* __restrict__ Q, const bf16* __restrict__ Kd,
    const bf16* __restrict__ Vt, const unsigned int* __restrict__ maskbits,
    bf16* __restrict__ Y) {
  __shared__ bf16 plds[4][16 * 32];  // per-wave P tile (row-major 16x32)
  int warp = threadIdx.x >> 5, lane = threadIdx.x & 31;
  int wave = blockIdx.x * 4 + warp;
  int qt = wave & 127;          // T/16 tiles
  int h = (wave >> 7) & 7;
  int b = wave >> 10;
  int q0 = qt * 16;

  const bf16* Qp = Q + ((size_t)(b * Hn + h)) * Tn * Dn;
  const bf16* Kp = Kd + ((size_t)(b * Hn + h)) * Tn * Dn;
  const bf16* Vp = Vt + ((size_t)(b * Hn + h)) * Dn * Tn;
  const unsigned int* Mp = maskbits + ((size_t)b * Tn + q0) * MW;

  int n = lane & 15, hf = lane >> 4;
  Frag qa0, qa1;
  load_a_frag(qa0, Qp, q0, Dn, 0, lane);
  load_a_frag(qa1, Qp, q0, Dn, 32, lane);

  v8f oacc[4] = {{}, {}, {}, {}};
  float mrow[8], lrow[8];   // lrow: per-LANE partial exp-sums (reduced once at end)
#pragma unroll
  for (int r = 0; r < 8; ++r) { mrow[r] = -3.0e38f; lrow[r] = 0.f; }

  bf16* pl = plds[warp];

  for (int kc = 0; kc < Tn / 32; ++kc) {
    int kb = kc * 32;
    // prefetch next chunk's K rows / V columns into cache while we compute
    if (kc + 1 < Tn / 32) {
      __builtin_prefetch(Kp + (size_t)(kb + 32 + lane) * Dn, 0, 1);
      __builtin_prefetch(Vp + (size_t)(lane & 63) * Tn + kb + 32, 0, 1);
    }

    // scores: S = Q Kt  (GEMM K-dim = D = 64 -> 2 WMMAs per 16-key subtile)
    Frag b0a, b0b, b1a, b1b;
    load_b_frag(b0a, Kp, kb, Dn, 0, lane);
    load_b_frag(b0b, Kp, kb, Dn, 32, lane);
    load_b_frag(b1a, Kp, kb + 16, Dn, 0, lane);
    load_b_frag(b1b, Kp, kb + 16, Dn, 32, lane);
    v8f s0 = {}, s1 = {};
    s0 = wmma_bf16(qa0, b0a, s0); s0 = wmma_bf16(qa1, b0b, s0);
    s1 = wmma_bf16(qa0, b1a, s1); s1 = wmma_bf16(qa1, b1b, s1);

    // mask + online softmax (per-lane VGPR r <-> row r+8*hf; cols across 16-lane half)
    unsigned int wbits[8];
#pragma unroll
    for (int r = 0; r < 8; ++r) wbits[r] = Mp[(size_t)(r + 8 * hf) * MW + kc];

#pragma unroll
    for (int r = 0; r < 8; ++r) {
      bool m0 = (wbits[r] >> n) & 1u;
      bool m1 = (wbits[r] >> (n + 16)) & 1u;
      float t0 = m0 ? s0[r] : -3.0e38f;
      float t1 = m1 ? s1[r] : -3.0e38f;
      float mx = fmaxf(t0, t1);
#pragma unroll
      for (int off = 1; off < 16; off <<= 1) mx = fmaxf(mx, __shfl_xor(mx, off, 32));
      float mnew = fmaxf(mrow[r], mx);         // uniform across each 16-lane half
      float alpha = __expf(mrow[r] - mnew);
      mrow[r] = mnew;
      float e0 = m0 ? __expf(s0[r] - mnew) : 0.f;
      float e1 = m1 ? __expf(s1[r] - mnew) : 0.f;
      lrow[r] = lrow[r] * alpha + e0 + e1;     // per-lane partial; reduce at end
      oacc[0][r] *= alpha; oacc[1][r] *= alpha;
      oacc[2][r] *= alpha; oacc[3][r] *= alpha;
      int row = r + 8 * hf;
      pl[row * 32 + n] = (bf16)e0;
      pl[row * 32 + 16 + n] = (bf16)e1;
    }
    asm volatile("s_wait_dscnt 0" ::: "memory");  // LDS layout shuffle fence (same wave)

    Frag pa;  // P as A-operand 16x32
    pa.u[0] = *(const uint4*)(pl + n * 32 + hf * 8);
    pa.u[1] = *(const uint4*)(pl + n * 32 + 16 + hf * 8);

    // O += P @ V   (B-operand from Vt[D][T]: fixed d row, contiguous keys)
#pragma unroll
    for (int dt = 0; dt < 4; ++dt) {
      Frag vb;
      load_b_frag(vb, Vp, dt * 16, Tn, kb, lane);
      oacc[dt] = wmma_bf16(pa, vb, oacc[dt]);
    }
  }

  // epilogue: reduce per-lane partial sums once, then O / l, write Y bf16
  bf16* Yp = Y + ((size_t)(b * Tn) + q0) * Cn + h * Dn;
#pragma unroll
  for (int r = 0; r < 8; ++r) {
    float ls = lrow[r];
#pragma unroll
    for (int off = 1; off < 16; off <<= 1) ls += __shfl_xor(ls, off, 32);
    float inv = 1.0f / ls;
    int row = r + 8 * hf;
#pragma unroll
    for (int dt = 0; dt < 4; ++dt)
      Yp[(size_t)row * Cn + dt * 16 + n] = (bf16)(oacc[dt][r] * inv);
  }
}

// ---------------- output projection ----------------
// 16x64 tile per wave, f32 output + bias
__global__ void __launch_bounds__(256) proj_gemm_kernel(
    const bf16* __restrict__ Yb, const bf16* __restrict__ wT,
    const float* __restrict__ bias, float* __restrict__ out) {
  int wave = (blockIdx.x * blockDim.x + threadIdx.x) >> 5;
  int lane = threadIdx.x & 31;
  const int NT = Cn / 64;  // 8 column tiles of 64
  int tm = wave / NT, tn = wave % NT;
  if (tm >= (Bn * Tn) / 16) return;

  v8f acc[4] = {{}, {}, {}, {}};
  for (int k0 = 0; k0 < Cn; k0 += 32) {
    Frag a;
    load_a_frag(a, Yb, tm * 16, Cn, k0, lane);
#pragma unroll
    for (int dt = 0; dt < 4; ++dt) {
      Frag bf;
      load_b_frag(bf, wT, tn * 64 + dt * 16, Cn, k0, lane);
      acc[dt] = wmma_bf16(a, bf, acc[dt]);
    }
  }

  int n = lane & 15, hf = lane >> 4;
#pragma unroll
  for (int dt = 0; dt < 4; ++dt) {
    int col = tn * 64 + dt * 16 + n;
    float bv = bias[col];
#pragma unroll
    for (int r = 0; r < 8; ++r) {
      int rowg = tm * 16 + r + 8 * hf;
      out[(size_t)rowg * Cn + col] = acc[dt][r] + bv;
    }
  }
}

// ---------------- host launcher ----------------
extern "C" void kernel_launch(void* const* d_in, const int* in_sizes, int n_in,
                              void* d_out, int out_size, void* d_ws, size_t ws_size,
                              hipStream_t stream) {
  const float* x = (const float*)d_in[0];
  const unsigned char* adj = (const unsigned char*)d_in[1];
  const float* w_qkv = (const float*)d_in[2];
  const float* b_qkv = (const float*)d_in[3];
  const float* w_proj = (const float*)d_in[4];
  const float* b_proj = (const float*)d_in[5];
  float* out = (float*)d_out;

  char* ws = (char*)d_ws;
  // workspace layout (all offsets >= 32B aligned), total ~44 MB
  bf16* xb     = (bf16*)(ws + 0);           //  8,388,608 B
  bf16* wqkvT  = (bf16*)(ws + 8388608);     //  1,572,864 B
  bf16* wprojT = (bf16*)(ws + 9961472);     //    524,288 B
  bf16* Qb     = (bf16*)(ws + 10485760);    //  8,388,608 B
  bf16* Kb     = (bf16*)(ws + 18874368);    //  8,388,608 B
  bf16* Vtb    = (bf16*)(ws + 27262976);    //  8,388,608 B
  bf16* Yb     = (bf16*)(ws + 35651584);    //  8,388,608 B
  unsigned int* mb = (unsigned int*)(ws + 44040192);  // 2,097,152 B

  const int nx = Bn * Tn * Cn;                 // 4,194,304
  cvt_f32_bf16_kernel<<<(nx + 255) / 256, 256, 0, stream>>>(x, xb, nx);
  transpose_w_kernel<<<(Cn * 3 * Cn + 255) / 256, 256, 0, stream>>>(w_qkv, wqkvT, Cn, 3 * Cn);
  transpose_w_kernel<<<(Cn * Cn + 255) / 256, 256, 0, stream>>>(w_proj, wprojT, Cn, Cn);
  const int nwords = Bn * Tn * MW;             // 524,288
  pack_mask_kernel<<<(nwords + 255) / 256, 256, 0, stream>>>(adj, mb, nwords);

  // QKV GEMM: (B*T/16) * (3C/64) = 512*24 waves, 8 waves/block
  qkv_gemm_kernel<<<(512 * 24) / 8, 256, 0, stream>>>(xb, wqkvT, b_qkv, Qb, Kb, Vtb);
  // attention: B*H*(T/16) = 4096 waves, 4 waves/block
  attn_kernel<<<4096 / 4, 128, 0, stream>>>(Qb, Kb, Vtb, mb, Yb);
  // output projection: (B*T/16) * (C/64) = 512*8 waves, 8 waves/block
  proj_gemm_kernel<<<(512 * 8) / 8, 256, 0, stream>>>(Yb, wprojT, b_proj, out);
}
